// PointNet2Encoder_75952201663080
// MI455X (gfx1250) — matrix operand order repaired and checked by reading, not compiled, not run on hardware
//
#include <hip/hip_runtime.h>
#include <hip/hip_bf16.h>

typedef __attribute__((ext_vector_type(16))) _Float16 v16h;
typedef __attribute__((ext_vector_type(8)))  _Float16 v8h;
typedef __attribute__((ext_vector_type(8)))  float    v8f;

#define BATCH 16
#define NSAMPLE 32
#define BN_EPS 1e-3f

static __host__ __device__ inline unsigned cdivu(size_t a, size_t b) { return (unsigned)((a + b - 1) / b); }

// --- CDNA5 async global->LDS copy helpers -----------------------------------
__device__ inline void async_copy_b128(unsigned lds_off, const void* gaddr)
{
    // GLOBAL_LOAD_ASYNC_TO_LDS_B128: vdst = per-lane LDS byte address,
    // vaddr = per-lane 64-bit global address, tracked by ASYNCcnt.
    asm volatile("global_load_async_to_lds_b128 %0, %1, off"
                 :: "v"(lds_off), "v"((unsigned long long)(uintptr_t)gaddr)
                 : "memory");
}

__device__ inline void wait_async0()
{
#if __has_builtin(__builtin_amdgcn_s_wait_asynccnt)
    __builtin_amdgcn_s_wait_asynccnt(0);
#else
    asm volatile("s_wait_asynccnt 0" ::: "memory");
#endif
}

// ---------------------------------------------------------------------------
// FPS: one block per batch. dist array lives in LDS; block-wide argmax reduce.
// ---------------------------------------------------------------------------
__global__ void fps_kernel(const float* __restrict__ xyz, int n, int npoint,
                           int* __restrict__ out)
{
    __shared__ float s_dist[4096];
    __shared__ float s_rv[256];
    __shared__ int   s_ri[256];
    __shared__ int   s_cur;
    const int b = blockIdx.x;
    const float* px = xyz + (size_t)b * n * 3;
    int* ob = out + (size_t)b * npoint;

    for (int i = threadIdx.x; i < n; i += blockDim.x) s_dist[i] = 1e10f;
    if (threadIdx.x == 0) s_cur = 0;
    __syncthreads();

    for (int s = 0; s < npoint; ++s) {
        const int cur = s_cur;
        if (threadIdx.x == 0) ob[s] = cur;
        const float cx = px[cur * 3 + 0];
        const float cy = px[cur * 3 + 1];
        const float cz = px[cur * 3 + 2];
        float bv = -1.0f; int bi = 0;
        for (int i = threadIdx.x; i < n; i += blockDim.x) {
            const float dx = px[i * 3 + 0] - cx;
            const float dy = px[i * 3 + 1] - cy;
            const float dz = px[i * 3 + 2] - cz;
            const float d  = dx * dx + dy * dy + dz * dz;
            const float nd = fminf(s_dist[i], d);
            s_dist[i] = nd;
            if (nd > bv) { bv = nd; bi = i; }
        }
        s_rv[threadIdx.x] = bv;
        s_ri[threadIdx.x] = bi;
        __syncthreads();
        for (int t = 128; t > 0; t >>= 1) {
            if ((int)threadIdx.x < t) {
                if (s_rv[threadIdx.x + t] > s_rv[threadIdx.x]) {
                    s_rv[threadIdx.x] = s_rv[threadIdx.x + t];
                    s_ri[threadIdx.x] = s_ri[threadIdx.x + t];
                }
            }
            __syncthreads();
        }
        if (threadIdx.x == 0) s_cur = s_ri[0];
        __syncthreads();
    }
}

// ---------------------------------------------------------------------------
// Gather sampled centers: new_xyz[b,m,:] = xyz[b, fps_idx[b,m], :]
// ---------------------------------------------------------------------------
__global__ void gather_xyz_kernel(const float* __restrict__ xyz,
                                  const int* __restrict__ fps_idx,
                                  float* __restrict__ new_xyz, int np, int n)
{
    const size_t total = (size_t)BATCH * np * 3;
    const size_t stride = (size_t)gridDim.x * blockDim.x;
    for (size_t i = (size_t)blockIdx.x * blockDim.x + threadIdx.x; i < total; i += stride) {
        const int c = (int)(i % 3);
        const size_t bm = i / 3;
        const int m = (int)(bm % np);
        const int b = (int)(bm / np);
        const int pi = fps_idx[(size_t)b * np + m];
        new_xyz[i] = xyz[((size_t)b * n + pi) * 3 + c];
    }
}

// ---------------------------------------------------------------------------
// Ball query: one wave32 per query point; nsample==32==wave width.
// ---------------------------------------------------------------------------
__global__ void ball_query_kernel(const float* __restrict__ new_xyz,
                                  const float* __restrict__ xyz,
                                  int* __restrict__ out_idx,
                                  int n, int np, float r2)
{
    const int gq   = blockIdx.x * 8 + (threadIdx.x >> 5);
    const int lane = threadIdx.x & 31;
    const int b = gq / np;
    const int m = gq % np;
    const float cx = new_xyz[((size_t)b * np + m) * 3 + 0];
    const float cy = new_xyz[((size_t)b * np + m) * 3 + 1];
    const float cz = new_xyz[((size_t)b * np + m) * 3 + 2];
    int* out = out_idx + ((size_t)b * np + m) * NSAMPLE;

    int found = 0;
    for (int base = 0; base < n && found < NSAMPLE; base += 32) {
        const int i = base + lane;
        const float dx = xyz[((size_t)b * n + i) * 3 + 0] - cx;
        const float dy = xyz[((size_t)b * n + i) * 3 + 1] - cy;
        const float dz = xyz[((size_t)b * n + i) * 3 + 2] - cz;
        const bool inside = (dx * dx + dy * dy + dz * dz) <= r2;
        const unsigned mask = (unsigned)__ballot(inside);
        const int pos = found + __popc(mask & ((1u << lane) - 1u));
        if (inside && pos < NSAMPLE) out[pos] = i;
        found += __popc(mask);
    }
    __syncthreads();  // make stores visible for fill pass
    const int first = (found > 0) ? out[0] : (n - 1);
    if (lane >= found) out[lane] = first;
}

// ---------------------------------------------------------------------------
// Grouping: build f16 GEMM input rows [B*np*32, Kpad]
// ---------------------------------------------------------------------------
__global__ void group_kernel(const float* __restrict__ xyz,
                             const float* __restrict__ points,
                             const float* __restrict__ new_xyz,
                             const int* __restrict__ idx,
                             _Float16* __restrict__ Xout,
                             int n, int np, int Cprev, int Kpad)
{
    const size_t total = (size_t)BATCH * np * NSAMPLE * Kpad;
    const size_t stride = (size_t)gridDim.x * blockDim.x;
    for (size_t i = (size_t)blockIdx.x * blockDim.x + threadIdx.x; i < total; i += stride) {
        const int c = (int)(i % Kpad);
        const size_t row = i / Kpad;
        const int k = (int)(row % NSAMPLE);
        const size_t bm = row / NSAMPLE;
        const int b = (int)(bm / np);
        const int pi = idx[bm * NSAMPLE + k];
        float v = 0.0f;
        if (c < 3) {
            v = xyz[((size_t)b * n + pi) * 3 + c] - new_xyz[bm * 3 + c];
        } else if (points != nullptr && (c - 3) < Cprev) {
            v = points[((size_t)b * n + pi) * Cprev + (c - 3)];
        }
        Xout[i] = (_Float16)v;
    }
}

// ---------------------------------------------------------------------------
// Weight convert+transpose: W f32 [K,Nc] -> Wt f16 [Nc,Kpad] (zero K-pad).
// ---------------------------------------------------------------------------
__global__ void convert_wt_kernel(const float* __restrict__ W,
                                  _Float16* __restrict__ Wt,
                                  int K, int Nc, int Kpad)
{
    const size_t total = (size_t)Nc * Kpad;
    const size_t stride = (size_t)gridDim.x * blockDim.x;
    for (size_t i = (size_t)blockIdx.x * blockDim.x + threadIdx.x; i < total; i += stride) {
        const int k = (int)(i % Kpad);
        const int nn = (int)(i / Kpad);
        Wt[i] = (k < K) ? (_Float16)W[(size_t)k * Nc + nn] : (_Float16)0.0f;
    }
}

// ---------------------------------------------------------------------------
// WMMA GEMM with LDS-staged B and 2-way M-tile interleaving:
//  - block owns one 16-column slice of Wt (16 x K f16), async-copied
//    global->LDS once (ASYNCcnt), reused by 8 waves x 4 M-tiles.
//  - K-loop processes 2 M-tiles with independent accumulators: breaks the
//    WMMA->WMMA RAW chain and reuses each LDS B fragment twice.
// ---------------------------------------------------------------------------
#define TM_PER_WAVE 4
__global__ void wmma_gemm_kernel(const _Float16* __restrict__ X,
                                 const _Float16* __restrict__ Wt,
                                 const float* __restrict__ bias,
                                 float* __restrict__ Out,
                                 int M, int K, int Nc)
{
    __shared__ __align__(16) _Float16 sB[16 * 512];   // 16 cols x K (K<=512)
    const int wave = threadIdx.x >> 5;
    const int lane = threadIdx.x & 31;
    const int ntN = Nc >> 4;
    const int Mt = M >> 4;
    const int tn = blockIdx.x % ntN;
    const int tmBase = (blockIdx.x / ntN) * (8 * TM_PER_WAVE);

    // --- cooperative async copy of this block's B slice into LDS ---
    {
        const char* src = (const char*)(Wt + (size_t)tn * 16 * K);
        const unsigned lbase = (unsigned)(uintptr_t)(void*)sB;  // LDS byte addr
        const int nchunks = 2 * K;                              // 16B chunks
        for (int c = threadIdx.x; c < nchunks; c += 256)
            async_copy_b128(lbase + (unsigned)c * 16u, src + (size_t)c * 16);
    }
    wait_async0();
    __syncthreads();

    const int l16 = lane & 15;
    const int khi = (lane >> 4) << 3;          // 0/8  (A K-sub-block)
    const int kb  = (lane >> 4) << 4;          // 0/16 (unused; B is indexed with khi)
    const _Float16* wlds = sB + (size_t)l16 * K;
    const int colg = tn * 16 + l16;
    const float bv = bias ? bias[colg] : 0.0f;

    for (int i = 0; i < TM_PER_WAVE; i += 2) {
        const int tm0 = tmBase + wave * TM_PER_WAVE + i;
        if (tm0 >= Mt) continue;               // wave-uniform: EXEC stays full
        const int tm1 = tm0 + 1;
        const bool has1 = (tm1 < Mt);          // wave-uniform
        const int tm1c = has1 ? tm1 : tm0;     // clamp to a valid row
        const _Float16* xr0 = X + (size_t)(tm0 * 16 + l16) * K;
        const _Float16* xr1 = X + (size_t)(tm1c * 16 + l16) * K;

        v8f acc0 = {}, acc1 = {};
        for (int kk = 0; kk < K; kk += 32) {
            if (kk + 32 < K) {
                __builtin_prefetch(xr0 + kk + 32, 0, 1);
                __builtin_prefetch(xr1 + kk + 32, 0, 1);
            }
            const v8h a00 = *(const v8h*)(xr0 + kk + khi);
            const v8h a01 = *(const v8h*)(xr0 + kk + 16 + khi);
            const v8h a10 = *(const v8h*)(xr1 + kk + khi);
            const v8h a11 = *(const v8h*)(xr1 + kk + 16 + khi);
            const v8h b0  = *(const v8h*)(wlds + kk + khi);
            const v8h b1  = *(const v8h*)(wlds + kk + 16 + khi);
            v16h a0, a1, bb;
#pragma unroll
            for (int j = 0; j < 8; ++j) {
                a0[j] = a00[j]; a0[j + 8] = a01[j];
                a1[j] = a10[j]; a1[j + 8] = a11[j];
                bb[j] = b0[j];  bb[j + 8] = b1[j];
            }
            acc0 = __builtin_amdgcn_wmma_f32_16x16x32_f16(
                false, a0, false, bb, (short)0, acc0, false, false);
            acc1 = __builtin_amdgcn_wmma_f32_16x16x32_f16(
                false, a1, false, bb, (short)0, acc1, false, false);
        }

        const int r0 = tm0 * 16 + ((lane >> 4) << 3);
#pragma unroll
        for (int r = 0; r < 8; ++r)
            Out[(size_t)(r0 + r) * Nc + colg] = acc0[r] + bv;
        if (has1) {
            const int r1 = tm1 * 16 + ((lane >> 4) << 3);
#pragma unroll
            for (int r = 0; r < 8; ++r)
                Out[(size_t)(r1 + r) * Nc + colg] = acc1[r] + bv;
        }
    }
}

// ---------------------------------------------------------------------------
// BN helpers
// ---------------------------------------------------------------------------
__global__ void zero_kernel(float* __restrict__ p, int n)
{
    const int i = blockIdx.x * blockDim.x + threadIdx.x;
    if (i < n) p[i] = 0.0f;
}

__global__ void bn_stats_kernel(const float* __restrict__ H,
                                float* __restrict__ sums,
                                float* __restrict__ sumsq, int M, int C)
{
    for (int c = threadIdx.x; c < C; c += blockDim.x) {
        float s = 0.0f, s2 = 0.0f;
        for (int r = blockIdx.x; r < M; r += gridDim.x) {
            const float x = H[(size_t)r * C + c];
            s += x; s2 += x * x;
        }
        atomicAdd(&sums[c], s);
        atomicAdd(&sumsq[c], s2);
    }
}

__global__ void bn_apply_kernel(float* __restrict__ H,
                                _Float16* __restrict__ Xnext,
                                const float* __restrict__ sums,
                                const float* __restrict__ sumsq,
                                const float* __restrict__ gamma,
                                const float* __restrict__ beta,
                                int M, int C)
{
    const size_t total = (size_t)M * C;
    const size_t stride = (size_t)gridDim.x * blockDim.x;
    const float inv = 1.0f / (float)M;
    for (size_t i = (size_t)blockIdx.x * blockDim.x + threadIdx.x; i < total; i += stride) {
        const int c = (int)(i % C);
        const float mean = sums[c] * inv;
        const float var  = sumsq[c] * inv - mean * mean;
        const float sc = gamma[c] * rsqrtf(var + BN_EPS);
        float y = sc * (H[i] - mean) + beta[c];
        y = fmaxf(y, 0.0f);
        H[i] = y;
        if (Xnext) Xnext[i] = (_Float16)y;
    }
}

// ---------------------------------------------------------------------------
// Max-pool over nsample
// ---------------------------------------------------------------------------
__global__ void maxpool_kernel(const float* __restrict__ H,
                               float* __restrict__ out, int np, int C)
{
    const size_t total = (size_t)BATCH * np * C;
    const size_t stride = (size_t)gridDim.x * blockDim.x;
    for (size_t i = (size_t)blockIdx.x * blockDim.x + threadIdx.x; i < total; i += stride) {
        const int c = (int)(i % C);
        const size_t bm = i / C;
        float v = -1e30f;
        for (int k = 0; k < NSAMPLE; ++k)
            v = fmaxf(v, H[(bm * NSAMPLE + k) * (size_t)C + c]);
        out[i] = v;
    }
}

__global__ void global_max_kernel(const float* __restrict__ pts,
                                  _Float16* __restrict__ latf16, int np, int C)
{
    const size_t total = (size_t)BATCH * C;
    const size_t stride = (size_t)gridDim.x * blockDim.x;
    for (size_t i = (size_t)blockIdx.x * blockDim.x + threadIdx.x; i < total; i += stride) {
        const int c = (int)(i % C);
        const int b = (int)(i / C);
        float v = -1e30f;
        for (int m = 0; m < np; ++m)
            v = fmaxf(v, pts[((size_t)b * np + m) * C + c]);
        latf16[i] = (_Float16)v;
    }
}

__global__ void bn_fc_kernel(const float* __restrict__ H,
                             _Float16* __restrict__ Xnext,
                             const float* __restrict__ gamma,
                             const float* __restrict__ beta, int C)
{
    const int c = blockIdx.x * blockDim.x + threadIdx.x;
    if (c >= C) return;
    float s = 0.0f, s2 = 0.0f;
    for (int r = 0; r < BATCH; ++r) {
        const float x = H[(size_t)r * C + c];
        s += x; s2 += x * x;
    }
    const float mean = s / (float)BATCH;
    const float var  = s2 / (float)BATCH - mean * mean;
    const float sc = gamma[c] * rsqrtf(var + BN_EPS);
    for (int r = 0; r < BATCH; ++r) {
        float y = sc * (H[(size_t)r * C + c] - mean) + beta[c];
        y = fmaxf(y, 0.0f);
        Xnext[(size_t)r * C + c] = (_Float16)y;
    }
}

// ===========================================================================
extern "C" void kernel_launch(void* const* d_in, const int* in_sizes, int n_in,
                              void* d_out, int out_size, void* d_ws, size_t ws_size,
                              hipStream_t stream)
{
    (void)in_sizes; (void)n_in; (void)out_size; (void)ws_size;
    // --- parameter mapping (JAX tree flatten: sorted dict keys) ---
    const float* x     = (const float*)d_in[0];
    const float* fc1W  = (const float*)d_in[1];
    const float* fc1g  = (const float*)d_in[3];
    const float* fc1be = (const float*)d_in[4];
    const float* fc2W  = (const float*)d_in[5];
    const float* fc2g  = (const float*)d_in[7];
    const float* fc2be = (const float*)d_in[8];
    const float* fc3W  = (const float*)d_in[9];
    const float* fc3b  = (const float*)d_in[10];
    const float *saW[4][3], *saG[4][3], *saBe[4][3];
    for (int l = 0; l < 4; ++l)
        for (int j = 0; j < 3; ++j) {
            const int base = 11 + l * 12 + j * 4;
            saW[l][j]  = (const float*)d_in[base + 0];
            saG[l][j]  = (const float*)d_in[base + 2];
            saBe[l][j] = (const float*)d_in[base + 3];
        }

    // --- level constants ---
    static const int   npv[4]    = {1024, 512, 256, 128};
    static const float r2v[4]    = {0.01f, 0.04f, 0.16f, 0.64f};
    static const int   cin[4]    = {3, 67, 131, 259};
    static const int   kpad0[4]  = {32, 96, 160, 288};
    static const int   cout[4][3]= {{32,32,64},{64,64,128},{128,128,256},{256,256,512}};

    // --- workspace layout ---
    char* ws = (char*)d_ws;
    size_t off = 0;
    auto alloc = [&](size_t bytes) -> char* {
        char* p = ws + off;
        off += (bytes + 255) & ~(size_t)255;
        return p;
    };
    int*   fpsIdx = (int*)alloc((size_t)BATCH * 1024 * 4);
    int*   bqIdx  = (int*)alloc((size_t)BATCH * 1024 * NSAMPLE * 4);
    float* nxyz[4];
    float* pts[4];
    for (int l = 0; l < 4; ++l) nxyz[l] = (float*)alloc((size_t)BATCH * npv[l] * 3 * 4);
    for (int l = 0; l < 4; ++l) pts[l]  = (float*)alloc((size_t)BATCH * npv[l] * cout[l][2] * 4);
    _Float16* bufA = (_Float16*)alloc(52428800);      // grouped input / L2 output
    _Float16* bufB = (_Float16*)alloc(33554432);      // L1 output
    float*    Hbuf = (float*)alloc(134217728);        // f32 activations
    _Float16* Wt   = (_Float16*)alloc(512 * 1024);
    float*    sums = (float*)alloc(512 * 4);
    float*    sumsq= (float*)alloc(512 * 4);
    float*    fcH  = (float*)alloc((size_t)BATCH * 512 * 4);
    _Float16* fcX1 = (_Float16*)alloc((size_t)BATCH * 512 * 2);
    _Float16* fcX2 = (_Float16*)alloc((size_t)BATCH * 512 * 2);

    auto gs = [](size_t total) -> unsigned {
        unsigned g = cdivu(total, 256);
        return g > 16384u ? 16384u : (g == 0u ? 1u : g);
    };
    auto gemm_grid = [](int M, int Nc) -> unsigned {
        const int Mt = M / 16, ntN = Nc / 16;
        return (unsigned)(ntN * ((Mt + (8 * TM_PER_WAVE) - 1) / (8 * TM_PER_WAVE)));
    };

    // --- SA stages ---
    for (int l = 0; l < 4; ++l) {
        const int Nl = (l == 0) ? 4096 : npv[l - 1];
        const int np = npv[l];
        const int Cprev = (l == 0) ? 0 : cout[l - 1][2];
        const float* xyz_l = (l == 0) ? x : nxyz[l - 1];
        const float* pts_l = (l == 0) ? nullptr : pts[l - 1];
        const int M = BATCH * np * NSAMPLE;

        fps_kernel<<<BATCH, 256, 0, stream>>>(xyz_l, Nl, np, fpsIdx);
        gather_xyz_kernel<<<gs((size_t)BATCH * np * 3), 256, 0, stream>>>(
            xyz_l, fpsIdx, nxyz[l], np, Nl);
        ball_query_kernel<<<(BATCH * np) / 8, 256, 0, stream>>>(
            nxyz[l], xyz_l, bqIdx, Nl, np, r2v[l]);
        group_kernel<<<gs((size_t)M * kpad0[l]), 256, 0, stream>>>(
            xyz_l, pts_l, nxyz[l], bqIdx, bufA, Nl, np, Cprev, kpad0[l]);

        _Float16* curin = bufA;
        int K = kpad0[l];
        for (int j = 0; j < 3; ++j) {
            const int Kreal = (j == 0) ? cin[l] : cout[l][j - 1];
            const int Nc = cout[l][j];
            convert_wt_kernel<<<gs((size_t)Nc * K), 256, 0, stream>>>(
                saW[l][j], Wt, Kreal, Nc, K);
            wmma_gemm_kernel<<<gemm_grid(M, Nc), 256, 0, stream>>>(
                curin, Wt, nullptr, Hbuf, M, K, Nc);
            zero_kernel<<<4, 256, 0, stream>>>(sums, 512);
            zero_kernel<<<4, 256, 0, stream>>>(sumsq, 512);
            bn_stats_kernel<<<256, 256, 0, stream>>>(Hbuf, sums, sumsq, M, Nc);
            _Float16* nxt = (j == 2) ? nullptr : ((curin == bufA) ? bufB : bufA);
            bn_apply_kernel<<<gs((size_t)M * Nc), 256, 0, stream>>>(
                Hbuf, nxt, sums, sumsq, saG[l][j], saBe[l][j], M, Nc);
            curin = nxt;
            K = Nc;
        }
        maxpool_kernel<<<gs((size_t)BATCH * np * cout[l][2]), 256, 0, stream>>>(
            Hbuf, pts[l], np, cout[l][2]);
    }

    // --- head ---
    global_max_kernel<<<gs((size_t)BATCH * 512), 256, 0, stream>>>(pts[3], fcX1, npv[3], 512);

    convert_wt_kernel<<<gs(512 * 256), 256, 0, stream>>>(fc1W, Wt, 512, 256, 512);
    wmma_gemm_kernel<<<gemm_grid(16, 256), 256, 0, stream>>>(fcX1, Wt, nullptr, fcH, 16, 512, 256);
    bn_fc_kernel<<<1, 256, 0, stream>>>(fcH, fcX2, fc1g, fc1be, 256);

    convert_wt_kernel<<<gs(256 * 256), 256, 0, stream>>>(fc2W, Wt, 256, 256, 256);
    wmma_gemm_kernel<<<gemm_grid(16, 256), 256, 0, stream>>>(fcX2, Wt, nullptr, fcH, 16, 256, 256);
    bn_fc_kernel<<<1, 256, 0, stream>>>(fcH, fcX1, fc2g, fc2be, 256);

    convert_wt_kernel<<<gs(256 * 512), 256, 0, stream>>>(fc3W, Wt, 256, 512, 256);
    wmma_gemm_kernel<<<gemm_grid(16, 512), 256, 0, stream>>>(fcX1, Wt, fc3b, (float*)d_out, 16, 256, 512);
}
